// SelfAttention_13804024889565
// MI455X (gfx1250) — matrix operand order repaired and checked by reading, not compiled, not run on hardware
//
#include <hip/hip_runtime.h>
#include <hip/hip_bf16.h>

typedef __attribute__((ext_vector_type(16))) __bf16 v16bf;
typedef __attribute__((ext_vector_type(8)))  __bf16 v8bf;
typedef __attribute__((ext_vector_type(8)))  float  v8f;

#define NEG_INF_F (-1.0e9f)

constexpr int N_TOK  = 4096;
constexpr int D_MODEL = 256;
constexpr int NH     = 8;
constexpr int HD     = 32;

static __device__ inline v8f wmma_bf16(v16bf a, v16bf b, v8f c) {
  return __builtin_amdgcn_wmma_f32_16x16x32_bf16(
      /*neg_a=*/false, a, /*neg_b=*/false, b,
      /*c_mod=*/(short)0, c, /*reuse_a=*/false, /*reuse_b=*/false);
}

// A-matrix 16x32 bf16 per ISA layout:
// lanes 0-15: M=lane, K=0..7 (elems 0-7) and K=16..23 (elems 8-15)
// lanes 16-31: M=lane-16, K=8..15 and K=24..31
static __device__ inline v16bf load_a16x32(const __bf16* src, int ld, int lane) {
  int half = lane >> 4, m = lane & 15;
  const v8bf* p0 = (const v8bf*)(src + m * ld + half * 8);
  const v8bf* p1 = (const v8bf*)(src + m * ld + 16 + half * 8);
  v8bf a = *p0, b = *p1;
  v16bf r;
#pragma unroll
  for (int i = 0; i < 8; ++i) { r[i] = a[i]; r[i + 8] = b[i]; }
  return r;
}

// B-matrix 32x16 bf16 per ISA layout: lane n holds column n,
// lanes 0-15: K=0..7 and K=16..23, lanes 16-31: K=8..15 and K=24..31.
// Source is stored so B's column n is contiguous: src + n*ld + k.
static __device__ inline v16bf load_b32x16(const __bf16* src, int ld, int lane) {
  int half = lane >> 4, n = lane & 15;
  const v8bf* p0 = (const v8bf*)(src + n * ld + half * 8);
  const v8bf* p1 = (const v8bf*)(src + n * ld + 16 + half * 8);
  v8bf a = *p0, b = *p1;
  v16bf r;
#pragma unroll
  for (int i = 0; i < 8; ++i) { r[i] = a[i]; r[i + 8] = b[i]; }
  return r;
}

// ---------------------------------------------------------------- kernel 0
__global__ void cvt_kernel(const float* __restrict__ x, const float* __restrict__ w_in,
                           const float* __restrict__ w_out,
                           __bf16* __restrict__ xb, __bf16* __restrict__ wbin,
                           __bf16* __restrict__ wbout) {
  int i = blockIdx.x * blockDim.x + threadIdx.x;
  if (i < N_TOK * D_MODEL)      xb[i]    = (__bf16)x[i];
  if (i < 3 * D_MODEL * D_MODEL) wbin[i] = (__bf16)w_in[i];
  if (i < D_MODEL * D_MODEL)    wbout[i] = (__bf16)w_out[i];
}

// ---------------------------------------------------------------- kernel 1
// qkv = x @ w_in^T + b_in  -> Qs (scaled) [h][n][d], Kb [h][n][d], Vt [h][d][n]
__global__ void qkv_kernel(const __bf16* __restrict__ xb, const __bf16* __restrict__ wbin,
                           const float* __restrict__ b_in,
                           __bf16* __restrict__ Qs, __bf16* __restrict__ Kb,
                           __bf16* __restrict__ Vt) {
  int wave = (blockIdx.x * blockDim.x + threadIdx.x) >> 5;
  int lane = threadIdx.x & 31;
  int rowTile = wave / 48, colTile = wave % 48;        // 768/16 = 48 col tiles
  int row0 = rowTile * 16, col0 = colTile * 16;

  v8f acc = {};
#pragma unroll
  for (int k = 0; k < D_MODEL; k += 32) {
    v16bf a = load_a16x32(xb + row0 * D_MODEL + k, D_MODEL, lane);
    v16bf b = load_b32x16(wbin + col0 * D_MODEL + k, D_MODEL, lane);
    acc = wmma_bf16(a, b, acc);
  }

  int n = lane & 15, half = lane >> 4;
  int c = col0 + n;
  float bias = b_in[c];

  if (c < D_MODEL) {                                   // Q (fold 1/sqrt(hd))
    int h = c >> 5, d = c & 31;
    const float scale = 0.17677669529663687f;
#pragma unroll
    for (int r = 0; r < 8; ++r) {
      int row = row0 + half * 8 + r;
      Qs[(h * N_TOK + row) * HD + d] = (__bf16)((acc[r] + bias) * scale);
    }
  } else if (c < 2 * D_MODEL) {                        // K
    int cc = c - D_MODEL; int h = cc >> 5, d = cc & 31;
#pragma unroll
    for (int r = 0; r < 8; ++r) {
      int row = row0 + half * 8 + r;
      Kb[(h * N_TOK + row) * HD + d] = (__bf16)(acc[r] + bias);
    }
  } else {                                             // V, transposed store
    int cc = c - 2 * D_MODEL; int h = cc >> 5, d = cc & 31;
    v8bf vv;
#pragma unroll
    for (int r = 0; r < 8; ++r) vv[r] = (__bf16)(acc[r] + bias);
    *(v8bf*)(Vt + (h * HD + d) * N_TOK + row0 + half * 8) = vv;
  }
}

// ---------------------------------------------------------------- kernel 2
// Flash attention, one wave per (head, 16-query tile), 32 keys per step.
__global__ void attn_kernel(const __bf16* __restrict__ Qs, const __bf16* __restrict__ Kb,
                            const __bf16* __restrict__ Vt, const int* __restrict__ bidx,
                            __bf16* __restrict__ attnb) {
  __shared__ __align__(32) __bf16 pbuf[4][16 * 32];
  int wid = threadIdx.x >> 5, lane = threadIdx.x & 31;
  int gwave = blockIdx.x * 4 + wid;
  int h = gwave >> 8, qt = gwave & 255;
  int q0 = qt * 16;
  int half = lane >> 4, n = lane & 15;

  v16bf aQ = load_a16x32(Qs + (h * N_TOK + q0) * HD, HD, lane);

  int qb[8];
#pragma unroll
  for (int r = 0; r < 8; ++r) qb[r] = bidx[q0 + half * 8 + r];
  int qbmin = bidx[q0], qbmax = bidx[q0 + 15];         // sorted -> tile range

  float mrow[8], lrow[8];
  v8f o0 = {}, o1 = {};
#pragma unroll
  for (int r = 0; r < 8; ++r) { mrow[r] = -0.5e9f; lrow[r] = 0.0f; }

  __bf16* myp = pbuf[wid];

  for (int j = 0; j < N_TOK; j += 32) {
    int kmin = bidx[j], kmax = bidx[j + 31];           // sorted -> block range
    if (kmax < qbmin || kmin > qbmax) continue;        // block-diagonal skip

    v16bf bK0 = load_b32x16(Kb + (h * N_TOK + j) * HD, HD, lane);
    v16bf bK1 = load_b32x16(Kb + (h * N_TOK + j + 16) * HD, HD, lane);
    v8f z = {};
    v8f s0 = wmma_bf16(aQ, bK0, z);
    v8f s1 = wmma_bf16(aQ, bK1, z);

    int kb0 = bidx[j + n], kb1 = bidx[j + 16 + n];
#pragma unroll
    for (int r = 0; r < 8; ++r) {
      float v0 = (qb[r] == kb0) ? s0[r] : NEG_INF_F;
      float v1 = (qb[r] == kb1) ? s1[r] : NEG_INF_F;
      float mx = fmaxf(v0, v1);
#pragma unroll
      for (int d = 1; d < 16; d <<= 1) mx = fmaxf(mx, __shfl_xor(mx, d, 32));
      float mn = fmaxf(mrow[r], mx);
      float al = __expf(mrow[r] - mn);
      float p0 = __expf(v0 - mn);
      float p1 = __expf(v1 - mn);
      float rs = p0 + p1;
#pragma unroll
      for (int d = 1; d < 16; d <<= 1) rs += __shfl_xor(rs, d, 32);
      lrow[r] = lrow[r] * al + rs;
      mrow[r] = mn;
      o0[r] *= al; o1[r] *= al;
      int rowi = half * 8 + r;
      myp[rowi * 32 + n]      = (__bf16)p0;            // C-layout -> LDS
      myp[rowi * 32 + 16 + n] = (__bf16)p1;
    }
    __asm__ volatile("s_wait_dscnt 0" ::: "memory");   // LDS transpose fence
    v16bf aP  = load_a16x32(myp, 32, lane);            // reload in A-layout
    v16bf bV0 = load_b32x16(Vt + (h * HD + 0) * N_TOK + j, N_TOK, lane);
    v16bf bV1 = load_b32x16(Vt + (h * HD + 16) * N_TOK + j, N_TOK, lane);
    o0 = wmma_bf16(aP, bV0, o0);
    o1 = wmma_bf16(aP, bV1, o1);
  }

#pragma unroll
  for (int r = 0; r < 8; ++r) {
    float inv = 1.0f / lrow[r];
    int row = q0 + half * 8 + r;
    attnb[row * D_MODEL + h * HD + n]      = (__bf16)(o0[r] * inv);
    attnb[row * D_MODEL + h * HD + 16 + n] = (__bf16)(o1[r] * inv);
  }
}

// ---------------------------------------------------------------- kernel 3
// z = x + attn @ w_out^T + b_out   (f32)
__global__ void proj_kernel(const __bf16* __restrict__ attnb, const __bf16* __restrict__ wbout,
                            const float* __restrict__ b_out, const float* __restrict__ x,
                            float* __restrict__ z) {
  int wave = (blockIdx.x * blockDim.x + threadIdx.x) >> 5;
  int lane = threadIdx.x & 31;
  int rowTile = wave >> 4, colTile = wave & 15;
  int row0 = rowTile * 16, col0 = colTile * 16;

  v8f acc = {};
#pragma unroll
  for (int k = 0; k < D_MODEL; k += 32) {
    v16bf a = load_a16x32(attnb + row0 * D_MODEL + k, D_MODEL, lane);
    v16bf b = load_b32x16(wbout + col0 * D_MODEL + k, D_MODEL, lane);
    acc = wmma_bf16(a, b, acc);
  }
  int n = lane & 15, half = lane >> 4;
  int col = col0 + n;
  float bias = b_out[col];
#pragma unroll
  for (int r = 0; r < 8; ++r) {
    int row = row0 + half * 8 + r;
    z[row * D_MODEL + col] = acc[r] + bias + x[row * D_MODEL + col];
  }
}

// ---------------------------------------------------------------- kernel 4
// out = layernorm(z) per row; one wave per row, 8 f32 per lane.
__global__ void ln_kernel(const float* __restrict__ z, const float* __restrict__ gamma,
                          const float* __restrict__ beta, float* __restrict__ out) {
  int wave = (blockIdx.x * blockDim.x + threadIdx.x) >> 5;
  int lane = threadIdx.x & 31;
  int row = wave;
  v8f v = *(const v8f*)(z + row * D_MODEL + lane * 8);
  float s = 0.0f;
#pragma unroll
  for (int i = 0; i < 8; ++i) s += v[i];
#pragma unroll
  for (int d = 1; d < 32; d <<= 1) s += __shfl_xor(s, d, 32);
  float mu = s * (1.0f / D_MODEL);
  float vs = 0.0f;
#pragma unroll
  for (int i = 0; i < 8; ++i) { float t = v[i] - mu; vs += t * t; }
#pragma unroll
  for (int d = 1; d < 32; d <<= 1) vs += __shfl_xor(vs, d, 32);
  float inv = rsqrtf(vs * (1.0f / D_MODEL) + 1e-5f);
#pragma unroll
  for (int i = 0; i < 8; ++i) {
    int c = lane * 8 + i;
    out[row * D_MODEL + c] = (v[i] - mu) * inv * gamma[c] + beta[c];
  }
}

// ---------------------------------------------------------------- launch
extern "C" void kernel_launch(void* const* d_in, const int* in_sizes, int n_in,
                              void* d_out, int out_size, void* d_ws, size_t ws_size,
                              hipStream_t stream) {
  const float* slots   = (const float*)d_in[0];
  const int*   bidx    = (const int*)d_in[1];
  const float* w_in    = (const float*)d_in[2];
  const float* b_in    = (const float*)d_in[3];
  const float* w_out   = (const float*)d_in[4];
  const float* b_out   = (const float*)d_in[5];
  const float* ln_g    = (const float*)d_in[6];
  const float* ln_b    = (const float*)d_in[7];
  float* out = (float*)d_out;

  // workspace carve-up (256B aligned)
  char* ws = (char*)d_ws;
  size_t off = 0;
  auto alloc = [&](size_t bytes) { char* p = ws + off; off = (off + bytes + 255) & ~size_t(255); return p; };
  __bf16* xb    = (__bf16*)alloc((size_t)N_TOK * D_MODEL * 2);
  __bf16* wbin  = (__bf16*)alloc((size_t)3 * D_MODEL * D_MODEL * 2);
  __bf16* wbout = (__bf16*)alloc((size_t)D_MODEL * D_MODEL * 2);
  __bf16* Qs    = (__bf16*)alloc((size_t)NH * N_TOK * HD * 2);
  __bf16* Kb    = (__bf16*)alloc((size_t)NH * N_TOK * HD * 2);
  __bf16* Vt    = (__bf16*)alloc((size_t)NH * HD * N_TOK * 2);
  __bf16* attnb = (__bf16*)alloc((size_t)N_TOK * D_MODEL * 2);
  float*  zbuf  = (float*)alloc((size_t)N_TOK * D_MODEL * 4);
  (void)ws_size;

  cvt_kernel<<<(N_TOK * D_MODEL + 255) / 256, 256, 0, stream>>>(slots, w_in, w_out, xb, wbin, wbout);
  qkv_kernel<<<(256 * 48) / 8, 256, 0, stream>>>(xb, wbin, b_in, Qs, Kb, Vt);
  attn_kernel<<<(NH * 256) / 4, 128, 0, stream>>>(Qs, Kb, Vt, bidx, attnb);
  proj_kernel<<<(256 * 16) / 8, 256, 0, stream>>>(attnb, wbout, b_out, slots, zbuf);
  ln_kernel<<<N_TOK / 8, 256, 0, stream>>>(zbuf, ln_g, ln_b, out);
}
